// RNN_40200893891032
// MI455X (gfx1250) — compile-verified, matrix operand order checked by reading, not run on hardware
//
#include <hip/hip_runtime.h>

// Problem constants (match reference)
#define BB 128
#define TT 2048
#define DD 64
#define HH 128
#define OO 32
#define SLICES 8          // B / 16
#define RB 16             // batch rows per block

typedef _Float16 h8v  __attribute__((ext_vector_type(8)));
typedef _Float16 h16v __attribute__((ext_vector_type(16)));
typedef float    f8v  __attribute__((ext_vector_type(8)));

// ---------------------------------------------------------------------------
// Fragment loader for 16-bit A/B operands of V_WMMA_*_16X16X32 (wave32).
// A (16xK, row-major, stride in halves): lane L holds row L&15, K-halves
// {ko..ko+7, ko+16..ko+23}, ko = kchunk*32 + (L&16 ? 8 : 0).
// B (KxN) uses the same routine on W stored row-major [N][K] (i.e. W^T).
// Two 16-byte loads -> ds_load_b128 / global_load_b128 after addrspace infer.
// ---------------------------------------------------------------------------
__device__ __forceinline__ h16v load_frag(const _Float16* base, int strideHalves, int kchunk) {
    const int lane = threadIdx.x & 31;
    const int row  = lane & 15;
    const int ko   = kchunk * 32 + ((lane & 16) ? 8 : 0);
    const h8v lo = *(const h8v*)(base + row * strideHalves + ko);
    const h8v hi = *(const h8v*)(base + row * strideHalves + ko + 16);
    h16v f;
#pragma unroll
    for (int i = 0; i < 8; ++i) { f[i] = lo[i]; f[i + 8] = hi[i]; }
    return f;
}

__device__ __forceinline__ f8v wmma16(h16v a, h16v b, f8v c) {
    // D = A*B + C ; f32 accumulate, f16 operands
    return __builtin_amdgcn_wmma_f32_16x16x32_f16(false, a, false, b, (short)0, c, false, false);
}

// ---------------------------------------------------------------------------
// Kernel 1: xp0 = x @ W_ih0^T + b_ih0 + b_hh0    (f32 in, f16 xp out)
// grid (8 slices, 8 t-chunks), 256 threads (8 waves), one t per wave per iter.
// ---------------------------------------------------------------------------
__global__ void __launch_bounds__(256) proj_x_kernel(
        const float* __restrict__ x,      // [B][T][64]
        const float* __restrict__ W,      // [128][64]
        const float* __restrict__ bih, const float* __restrict__ bhh,
        _Float16* __restrict__ xp)        // [8][T][16][128]
{
    __shared__ __align__(16) _Float16 lds_w[HH * DD];     // 16 KB
    __shared__ __align__(16) float    lds_b[HH];
    __shared__ __align__(16) _Float16 lds_x[8][RB * DD];  // 8 x 2 KB

    const int tid = threadIdx.x, lane = tid & 31, wv = tid >> 5;
    const int s  = blockIdx.x;
    const int t0 = blockIdx.y * 256;
    for (int i = tid; i < HH * DD; i += 256) lds_w[i] = (_Float16)W[i];
    for (int i = tid; i < HH; i += 256)      lds_b[i] = bih[i] + bhh[i];
    __syncthreads();

    const int col = lane & 15;
    const int hi8 = (lane & 16) ? 8 : 0;

    for (int t = t0 + wv; t < t0 + 256; t += 8) {
        // stage this wave's 16x64 x-tile as f16 (wave-local, no barrier needed)
        for (int i = lane; i < RB * DD; i += 32) {
            int r = i >> 6, c = i & 63;
            lds_x[wv][i] = (_Float16)x[((size_t)(s * RB + r) * TT + t) * DD + c];
        }
        const size_t obase = (((size_t)s * TT + t) * RB) * HH;
        for (int n = 0; n < 8; ++n) {
            f8v acc;
#pragma unroll
            for (int r = 0; r < 8; ++r) acc[r] = lds_b[n * 16 + col];
#pragma unroll
            for (int kc = 0; kc < 2; ++kc) {
                h16v a = load_frag(lds_x[wv], DD, kc);
                h16v b = load_frag(lds_w + n * 16 * DD, DD, kc);
                acc = wmma16(a, b, acc);
            }
#pragma unroll
            for (int r = 0; r < 8; ++r)
                xp[obase + (size_t)(r + hi8) * HH + n * 16 + col] = (_Float16)acc[r];
        }
    }
}

// ---------------------------------------------------------------------------
// Kernel 3: xp1 = h1 @ W_ih1^T + b_ih1 + b_hh1   (f16 in, f16 out, K=128)
// A-fragments load straight from global h-buffer (already f16, tile-contig).
// ---------------------------------------------------------------------------
__global__ void __launch_bounds__(256) proj_h_kernel(
        const _Float16* __restrict__ hin, // [8][T][16][128]
        const float* __restrict__ W,      // [128][128]
        const float* __restrict__ bih, const float* __restrict__ bhh,
        _Float16* __restrict__ xp)        // [8][T][16][128]
{
    __shared__ __align__(16) _Float16 lds_w[HH * HH];     // 32 KB
    __shared__ __align__(16) float    lds_b[HH];

    const int tid = threadIdx.x, lane = tid & 31, wv = tid >> 5;
    const int s  = blockIdx.x;
    const int t0 = blockIdx.y * 256;
    for (int i = tid; i < HH * HH; i += 256) lds_w[i] = (_Float16)W[i];
    for (int i = tid; i < HH; i += 256)      lds_b[i] = bih[i] + bhh[i];
    __syncthreads();

    const int col = lane & 15;
    const int hi8 = (lane & 16) ? 8 : 0;

    for (int t = t0 + wv; t < t0 + 256; t += 8) {
        const size_t tb = (((size_t)s * TT + t) * RB) * HH;
        const _Float16* at = hin + tb;
        for (int n = 0; n < 8; ++n) {
            f8v acc;
#pragma unroll
            for (int r = 0; r < 8; ++r) acc[r] = lds_b[n * 16 + col];
#pragma unroll
            for (int kc = 0; kc < 4; ++kc) {
                h16v a = load_frag(at, HH, kc);
                h16v b = load_frag(lds_w + n * 16 * HH, HH, kc);
                acc = wmma16(a, b, acc);
            }
#pragma unroll
            for (int r = 0; r < 8; ++r)
                xp[tb + (size_t)(r + hi8) * HH + n * 16 + col] = (_Float16)acc[r];
        }
    }
}

// ---------------------------------------------------------------------------
// Kernels 2 & 4: serial recurrence  h_t = tanh(xp_t + h_{t-1} @ W_hh^T)
// with packed-sequence freeze (t >= len holds state). One block per 16-row
// batch slice -> no inter-block sync. 8 waves; each wave owns one 16-col
// output tile: 4 chained WMMAs/step on K=128. W_hh fragments hoisted to
// registers; h state double-buffered in LDS; one barrier per step; next-step
// xp software-prefetched across the WMMA chain.
// zero_pad=0: store held state (layer-1 output / ys semantics).
// zero_pad=1: store zero at frozen steps (pad_packed semantics for FC).
// ---------------------------------------------------------------------------
__global__ void __launch_bounds__(256) rec_kernel(
        const _Float16* __restrict__ xp,   // [8][T][16][128] (biases included)
        const float* __restrict__ Whh,     // [128][128]
        const int* __restrict__ lengths,   // [B]
        _Float16* __restrict__ hout,       // [8][T][16][128]
        int zero_pad)
{
    __shared__ __align__(16) _Float16 lds_w[HH * HH];       // 32 KB
    __shared__ __align__(16) _Float16 lds_h[2][RB * HH];    // 8 KB, double buffer

    const int tid = threadIdx.x, lane = tid & 31, wv = tid >> 5;
    const int s = blockIdx.x;
    for (int i = tid; i < HH * HH; i += 256) lds_w[i] = (_Float16)Whh[i];
    for (int i = tid; i < 2 * RB * HH; i += 256) ((_Float16*)lds_h)[i] = (_Float16)0.0f;
    __syncthreads();

    // hoist W_hh B-fragments for this wave's N-tile into registers (whole loop)
    h16v bw[4];
#pragma unroll
    for (int kc = 0; kc < 4; ++kc) bw[kc] = load_frag(lds_w + wv * 16 * HH, HH, kc);

    const int col = lane & 15;
    const int hi8 = (lane & 16) ? 8 : 0;
    int   len_r[8];
    float hold[8];
#pragma unroll
    for (int r = 0; r < 8; ++r) {
        len_r[r] = lengths[s * RB + r + hi8];
        hold[r]  = 0.0f;
    }

    const size_t base = (size_t)s * TT * RB * HH;
    const int ooff = wv * 16 + col;

    // prefetch xp for t = 0
    float xq[8];
#pragma unroll
    for (int r = 0; r < 8; ++r)
        xq[r] = (float)xp[base + (size_t)(r + hi8) * HH + ooff];

    int p = 0;
    for (int t = 0; t < TT; ++t) {
        f8v acc;
#pragma unroll
        for (int r = 0; r < 8; ++r) acc[r] = xq[r];
#pragma unroll
        for (int kc = 0; kc < 4; ++kc) {
            h16v a = load_frag(lds_h[p], HH, kc);
            acc = wmma16(a, bw[kc], acc);
        }
        if (t + 1 < TT) {
            const _Float16* xn = xp + base + (size_t)(t + 1) * RB * HH;
#pragma unroll
            for (int r = 0; r < 8; ++r) xq[r] = (float)xn[(r + hi8) * HH + ooff];
        }
#pragma unroll
        for (int r = 0; r < 8; ++r) {
            float hn   = tanhf(acc[r]);
            bool  keep = t < len_r[r];
            hold[r]    = keep ? hn : hold[r];
            float sv   = zero_pad ? (keep ? hn : 0.0f) : hold[r];
            lds_h[p ^ 1][(r + hi8) * HH + ooff] = (_Float16)hold[r];
            hout[base + (size_t)t * RB * HH + (size_t)(r + hi8) * HH + ooff] = (_Float16)sv;
        }
        p ^= 1;
        __syncthreads();
    }
}

// ---------------------------------------------------------------------------
// Kernel 5: out = h2_masked @ W_fc^T + b_fc   (f32 output, N=32 -> 2 tiles)
// ---------------------------------------------------------------------------
__global__ void __launch_bounds__(256) fc_kernel(
        const _Float16* __restrict__ hin, // [8][T][16][128]
        const float* __restrict__ Wfc,    // [32][128]
        const float* __restrict__ bfc,    // [32]
        float* __restrict__ out)          // [B][T][32]
{
    __shared__ __align__(16) _Float16 lds_w[OO * HH];   // 8 KB
    __shared__ __align__(16) float    lds_b[OO];

    const int tid = threadIdx.x, lane = tid & 31, wv = tid >> 5;
    const int s  = blockIdx.x;
    const int t0 = blockIdx.y * 256;
    for (int i = tid; i < OO * HH; i += 256) lds_w[i] = (_Float16)Wfc[i];
    for (int i = tid; i < OO; i += 256)      lds_b[i] = bfc[i];
    __syncthreads();

    const int col = lane & 15;
    const int hi8 = (lane & 16) ? 8 : 0;

    for (int t = t0 + wv; t < t0 + 256; t += 8) {
        const _Float16* at = hin + (((size_t)s * TT + t) * RB) * HH;
        for (int n = 0; n < 2; ++n) {
            f8v acc;
#pragma unroll
            for (int r = 0; r < 8; ++r) acc[r] = lds_b[n * 16 + col];
#pragma unroll
            for (int kc = 0; kc < 4; ++kc) {
                h16v a = load_frag(at, HH, kc);
                h16v b = load_frag(lds_w + n * 16 * HH, HH, kc);
                acc = wmma16(a, b, acc);
            }
#pragma unroll
            for (int r = 0; r < 8; ++r) {
                int brow = s * RB + r + hi8;
                out[((size_t)brow * TT + t) * OO + n * 16 + col] = acc[r];
            }
        }
    }
}

// ---------------------------------------------------------------------------
// Launch: proj0 -> rec0 -> proj1 -> rec1 -> fc   (stream-ordered dependencies)
// Workspace: xp buffer (64 MB f16, reused for both layers) + h buffer (64 MB
// f16, holds h1 then masked h2). Total 128 MB of d_ws.
// ---------------------------------------------------------------------------
extern "C" void kernel_launch(void* const* d_in, const int* in_sizes, int n_in,
                              void* d_out, int out_size, void* d_ws, size_t ws_size,
                              hipStream_t stream) {
    const float* x      = (const float*)d_in[0];
    const int*   lens   = (const int*)  d_in[1];
    const float* W_ih0  = (const float*)d_in[2];
    const float* W_hh0  = (const float*)d_in[3];
    const float* b_ih0  = (const float*)d_in[4];
    const float* b_hh0  = (const float*)d_in[5];
    const float* W_ih1  = (const float*)d_in[6];
    const float* W_hh1  = (const float*)d_in[7];
    const float* b_ih1  = (const float*)d_in[8];
    const float* b_hh1  = (const float*)d_in[9];
    const float* W_fc   = (const float*)d_in[10];
    const float* b_fc   = (const float*)d_in[11];
    float* out = (float*)d_out;

    _Float16* xp = (_Float16*)d_ws;                         // 64 MB
    _Float16* hb = xp + (size_t)BB * TT * HH;               // next 64 MB

    dim3 blk(256);
    proj_x_kernel<<<dim3(SLICES, 8), blk, 0, stream>>>(x, W_ih0, b_ih0, b_hh0, xp);
    rec_kernel   <<<dim3(SLICES),    blk, 0, stream>>>(xp, W_hh0, lens, hb, 0);
    proj_h_kernel<<<dim3(SLICES, 8), blk, 0, stream>>>(hb, W_ih1, b_ih1, b_hh1, xp);
    rec_kernel   <<<dim3(SLICES),    blk, 0, stream>>>(xp, W_hh1, lens, hb, 1);
    fc_kernel    <<<dim3(SLICES, 8), blk, 0, stream>>>(hb, W_fc, b_fc, out);
}
